// Model_40785009443538
// MI455X (gfx1250) — matrix operand order repaired and checked by reading, not compiled, not run on hardware
//
#include <hip/hip_runtime.h>
#include <hip/hip_bf16.h>
#include <stdint.h>

// ---------------------------------------------------------------------------
// Problem constants (reference: B=16, S=512, D=1024, H=16, DK=64)
// ---------------------------------------------------------------------------
#define PB   16
#define PS   512
#define PD   1024
#define PH   16
#define PDK  64
#define PM   (PB * PS)      // 8192 rows of the activation matrix
#define PGN  (PH * PS)      // 8192 cols of the gating projection

#define TM   128            // block tile M
#define TN   128            // block tile N
#define TK   32             // K slab per stage (one WMMA K-step)

typedef __bf16 bf16;
typedef __attribute__((ext_vector_type(16))) __bf16 bf16x16;
typedef __attribute__((ext_vector_type(8)))  __bf16 bf16x8;
typedef __attribute__((ext_vector_type(8)))  float  f32x8;
typedef __attribute__((ext_vector_type(8)))  float  v8f;

// ---------------------------------------------------------------------------
// CDNA5 async global->LDS copy (ASYNCcnt-tracked), 16 bytes per lane.
// ldsoff = low 32 bits of the generic pointer (flat->LDS aperture keeps the
// workgroup-relative LDS byte offset in addr[31:0]).
// ---------------------------------------------------------------------------
__device__ __forceinline__ void async_ld128(uint32_t ldsoff, const void* gaddr) {
    asm volatile("global_load_async_to_lds_b128 %0, %1, off"
                 :: "v"(ldsoff), "v"(gaddr) : "memory");
}
__device__ __forceinline__ void wait_async_le4() {
    asm volatile("s_wait_asynccnt 0x4" ::: "memory");
}
__device__ __forceinline__ void wait_async_0() {
    asm volatile("s_wait_asynccnt 0x0" ::: "memory");
}
__device__ __forceinline__ uint32_t lds_off(const void* p) {
    return (uint32_t)(uintptr_t)p;
}

// ---------------------------------------------------------------------------
// Fragment builders (wave32 WMMA 16x16x32 bf16 layouts, CDNA5 ISA 7.12.2)
//   A 16x32 bf16: lane L -> row = L&15, K = (L>>4)*8 + {0..7} and +16+{0..7}
//   B 32x16 bf16: lane L -> col = L&15, K = (L>>4)*8 + {0..7} and +16+{0..7}
// B sources are stored transposed [N x K] row-major so loads are contiguous.
// ---------------------------------------------------------------------------
__device__ __forceinline__ bf16x16 pack_a(bf16x8 c0, bf16x8 c1) {
    bf16x16 f;
#pragma unroll
    for (int e = 0; e < 8; ++e) { f[e] = c0[e]; f[e + 8] = c1[e]; }
    return f;
}

__device__ __forceinline__ bf16x16 frag_a_bf16(const bf16* __restrict__ A, int lda,
                                               int row, int k0, int lane) {
    int kb = (lane >> 4) * 8;
    const bf16* p = A + (size_t)row * lda + k0 + kb;
    return pack_a(*(const bf16x8*)(p), *(const bf16x8*)(p + 16));
}

__device__ __forceinline__ bf16x16 frag_a_f32(const float* __restrict__ A, int lda,
                                              int row, int k0, int lane) {
    int kb = (lane >> 4) * 8;
    const float* p = A + (size_t)row * lda + k0 + kb;
    f32x8 c0 = *(const f32x8*)(p);
    f32x8 c1 = *(const f32x8*)(p + 16);
    bf16x16 f;
#pragma unroll
    for (int e = 0; e < 8; ++e) { f[e] = (bf16)c0[e]; f[e + 8] = (bf16)c1[e]; }
    return f;
}

__device__ __forceinline__ bf16x16 frag_b_bf16(const bf16* __restrict__ Wt, int ldb,
                                               int col, int k0, int lane) {
    int ko = (lane >> 4) * 8;
    const bf16* p = Wt + (size_t)col * ldb + k0 + ko;
    return pack_a(*(const bf16x8*)(p), *(const bf16x8*)(p + 16));
}

// LDS-resident variants (slab is [rows][TK] row-major, k0 already removed)
__device__ __forceinline__ bf16x16 frag_a_lds(const bf16* s, int row, int lane) {
    int kb = (lane >> 4) * 8;
    const bf16* p = s + row * TK + kb;
    return pack_a(*(const bf16x8*)(p), *(const bf16x8*)(p + 16));
}
__device__ __forceinline__ bf16x16 frag_b_lds(const bf16* s, int col, int lane) {
    int ko = (lane >> 4) * 8;
    const bf16* p = s + col * TK + ko;
    return pack_a(*(const bf16x8*)(p), *(const bf16x8*)(p + 16));
}

__device__ __forceinline__ v8f wmma_bf16(bf16x16 a, bf16x16 b, v8f c) {
    return __builtin_amdgcn_wmma_f32_16x16x32_bf16(
        /*neg_a=*/false, a, /*neg_b=*/false, b,
        /*c_mod=*/(short)0, c, /*reuse_a=*/false, /*reuse_b=*/false);
}

// ---------------------------------------------------------------------------
// Conversion kernels
// ---------------------------------------------------------------------------
__global__ void cvt_f32_bf16(const float* __restrict__ in, bf16* __restrict__ out, int n) {
    int i = blockIdx.x * blockDim.x + threadIdx.x;
    if (i < n) out[i] = (bf16)in[i];
}

// in: [K x N] fp32, out: [N x K] bf16 (coalesced writes, strided reads hit L2)
__global__ void transpose_cvt(const float* __restrict__ in, bf16* __restrict__ out,
                              int K, int N) {
    int i = blockIdx.x * blockDim.x + threadIdx.x;
    if (i < K * N) {
        int nn = i / K, kk = i - nn * K;
        out[i] = (bf16)in[(size_t)kk * N + nn];
    }
}

// ---------------------------------------------------------------------------
// Epilogue scatter helper
// MODE 0: bf16 scattered as [b,h,s,dk]   (q / k projections)
// MODE 1: bf16 scattered as [b,h,dk,s]   (v, stored transposed)
// MODE 2: f32  scattered as [b,h,s,j]    (gating -> attn region)
// MODE 3: f32  row-major [M x N]         (output projection)
// ---------------------------------------------------------------------------
template <int MODE>
__device__ __forceinline__ void store_tile(v8f acc, int tm, int tn, int lane,
                                           const float* __restrict__ bias,
                                           float* __restrict__ outF,
                                           bf16* __restrict__ outB) {
    int cc = tn + (lane & 15);
    int rbase = tm + (lane >> 4) * 8;
    float bv = bias ? bias[cc] : 0.0f;
#pragma unroll
    for (int j = 0; j < 8; ++j) {
        int r = rbase + j;
        float val = acc[j] + bv;
        if (MODE == 0) {
            int b = r >> 9, s = r & 511, h = cc >> 6, dk = cc & 63;
            outB[((((size_t)b * PH + h) * PS + s) * PDK) + dk] = (bf16)val;
        } else if (MODE == 1) {
            int b = r >> 9, s = r & 511, h = cc >> 6, dk = cc & 63;
            outB[((((size_t)b * PH + h) * PDK + dk) * PS) + s] = (bf16)val;
        } else if (MODE == 2) {
            int b = r >> 9, s = r & 511, h = cc >> 9, jj = cc & 511;
            outF[((((size_t)b * PH + h) * PS + s) * PS) + jj] = val;
        } else {
            outF[(size_t)r * PD + cc] = val;
        }
    }
}

// ---------------------------------------------------------------------------
// Projection GEMM: C[M,N] = A[M,K](bf16) @ Wt[N,K]^T (bf16) + bias
// 256 threads = 8 waves. Block tile 128x128, double-buffered async LDS
// staging of 128x32 A/B slabs. Wave tile 64x32 (4x2 WMMA accumulators).
// ---------------------------------------------------------------------------
template <int MODE>
__global__ void __launch_bounds__(256)
gemm_proj(const bf16* __restrict__ A, const bf16* __restrict__ Wt,
          const float* __restrict__ bias, float* __restrict__ outF,
          bf16* __restrict__ outB, int K) {
    __shared__ bf16 As[2][TM * TK];   // 8 KB each
    __shared__ bf16 Bs[2][TN * TK];   // 8 KB each

    const int tid  = threadIdx.x;
    const int lane = tid & 31;
    const int wave = tid >> 5;
    const int bm = blockIdx.y * TM;
    const int bn = blockIdx.x * TN;
    const int wm = (wave >> 2) * 64;          // 0 / 64
    const int wn = (wave & 3) * 32;           // 0 / 32 / 64 / 96

    // stage mapping: 256 threads x 32B -> 8KB slab; row = tid>>1, ko = (tid&1)*16
    const int srow = tid >> 1;
    const int sko  = (tid & 1) * 16;

    v8f acc[4][2] = {};

    // prologue: stage slab 0 into buffer 0 (4 async b128 per thread)
    {
        const bf16* ga = A  + (size_t)(bm + srow) * K + sko;
        const bf16* gb = Wt + (size_t)(bn + srow) * K + sko;
        uint32_t la = lds_off(&As[0][srow * TK + sko]);
        uint32_t lb = lds_off(&Bs[0][srow * TK + sko]);
        async_ld128(la,      ga);
        async_ld128(la + 16, ga + 8);
        async_ld128(lb,      gb);
        async_ld128(lb + 16, gb + 8);
    }

    for (int k0 = 0; k0 < K; k0 += TK) {
        const int buf  = (k0 >> 5) & 1;
        const bool more = (k0 + TK) < K;
        if (more) {
            const int kn = k0 + TK;
            const bf16* ga = A  + (size_t)(bm + srow) * K + kn + sko;
            const bf16* gb = Wt + (size_t)(bn + srow) * K + kn + sko;
            uint32_t la = lds_off(&As[buf ^ 1][srow * TK + sko]);
            uint32_t lb = lds_off(&Bs[buf ^ 1][srow * TK + sko]);
            async_ld128(la,      ga);
            async_ld128(la + 16, ga + 8);
            async_ld128(lb,      gb);
            async_ld128(lb + 16, gb + 8);
            wait_async_le4();       // current slab landed; next slab in flight
        } else {
            wait_async_0();
        }
        __syncthreads();

        bf16x16 af[4], bfm[2];
#pragma unroll
        for (int mt = 0; mt < 4; ++mt)
            af[mt] = frag_a_lds(&As[buf][0], wm + mt * 16 + (lane & 15), lane);
#pragma unroll
        for (int nt = 0; nt < 2; ++nt)
            bfm[nt] = frag_b_lds(&Bs[buf][0], wn + nt * 16 + (lane & 15), lane);
#pragma unroll
        for (int mt = 0; mt < 4; ++mt)
#pragma unroll
            for (int nt = 0; nt < 2; ++nt)
                acc[mt][nt] = wmma_bf16(af[mt], bfm[nt], acc[mt][nt]);

        __syncthreads();            // protect buf before it is re-staged
    }

#pragma unroll
    for (int mt = 0; mt < 4; ++mt)
#pragma unroll
        for (int nt = 0; nt < 2; ++nt)
            store_tile<MODE>(acc[mt][nt], bm + wm + mt * 16, bn + wn + nt * 16,
                             lane, bias, outF, outB);
}

// ---------------------------------------------------------------------------
// scores[b,h,i,j] = scale * (q[b,h,i,:] . k[b,h,j,:]) + prev[b,h,i,j]
// grid (S/64, S/64, B*H); K = DK = 64 (2 WMMA steps); direct register path.
// ---------------------------------------------------------------------------
__global__ void gemm_scores(const bf16* __restrict__ qbf, const bf16* __restrict__ kbf,
                            const float* __restrict__ prev, const float* __restrict__ scalep,
                            float* __restrict__ scores) {
    int bh = blockIdx.z;
    const bf16* Aq = qbf + (size_t)bh * PS * PDK;
    const bf16* Bk = kbf + (size_t)bh * PS * PDK;
    const float* Pv = prev + (size_t)bh * PS * PS;
    float* Sc = scores + (size_t)bh * PS * PS;
    float scale = scalep[0];

    int lane = threadIdx.x & 31;
    int wave = threadIdx.x >> 5;
    int wm = blockIdx.y * 64 + (wave >> 1) * 32;
    int wn = blockIdx.x * 64 + (wave & 1) * 32;
    int rA0 = wm + (lane & 15), rA1 = rA0 + 16;
    int cB0 = wn + (lane & 15), cB1 = cB0 + 16;
    v8f c00 = {}, c01 = {}, c10 = {}, c11 = {};
#pragma unroll
    for (int k0 = 0; k0 < PDK; k0 += 32) {
        bf16x16 a0 = frag_a_bf16(Aq, PDK, rA0, k0, lane);
        bf16x16 a1 = frag_a_bf16(Aq, PDK, rA1, k0, lane);
        bf16x16 b0 = frag_b_bf16(Bk, PDK, cB0, k0, lane);
        bf16x16 b1 = frag_b_bf16(Bk, PDK, cB1, k0, lane);
        c00 = wmma_bf16(a0, b0, c00);
        c01 = wmma_bf16(a0, b1, c01);
        c10 = wmma_bf16(a1, b0, c10);
        c11 = wmma_bf16(a1, b1, c11);
    }
    int cc = lane & 15;
    int rb = (lane >> 4) * 8;
    v8f accs[4] = {c00, c01, c10, c11};
    int tms[4] = {wm, wm, wm + 16, wm + 16};
    int tns[4] = {wn, wn + 16, wn, wn + 16};
#pragma unroll
    for (int t = 0; t < 4; ++t) {
#pragma unroll
        for (int j = 0; j < 8; ++j) {
            size_t idx = (size_t)(tms[t] + rb + j) * PS + (tns[t] + cc);
            Sc[idx] = accs[t][j] * scale + Pv[idx];
        }
    }
}

// ---------------------------------------------------------------------------
// Row softmax + gating: attn[row,:] = g[row,:] * softmax(scores[row,:])
// g pre-stored in the attn region by the gating GEMM; overwritten in place.
// ---------------------------------------------------------------------------
__global__ void softmax_gate(const float* __restrict__ scores, float* __restrict__ attn) {
    size_t row = blockIdx.x;
    const float* sr = scores + row * PS;
    float* ar = attn + row * PS;
    int t = threadIdx.x, lane = t & 31, w = t >> 5;

    float x0 = sr[t], x1 = sr[t + 256];
    float m = fmaxf(x0, x1);
#pragma unroll
    for (int off = 16; off > 0; off >>= 1) m = fmaxf(m, __shfl_xor(m, off, 32));
    __shared__ float redm[8];
    __shared__ float reds[8];
    if (lane == 0) redm[w] = m;
    __syncthreads();
    m = redm[0];
#pragma unroll
    for (int i = 1; i < 8; ++i) m = fmaxf(m, redm[i]);

    float e0 = __expf(x0 - m), e1 = __expf(x1 - m);
    float s = e0 + e1;
#pragma unroll
    for (int off = 16; off > 0; off >>= 1) s += __shfl_xor(s, off, 32);
    if (lane == 0) reds[w] = s;
    __syncthreads();
    s = reds[0];
#pragma unroll
    for (int i = 1; i < 8; ++i) s += reds[i];
    float inv = 1.0f / s;

    float g0 = ar[t], g1 = ar[t + 256];
    ar[t] = g0 * e0 * inv;
    ar[t + 256] = g1 * e1 * inv;
}

// ---------------------------------------------------------------------------
// Y[b,s,h*64+dk] = sum_j attn[b,h,s,j] * v[b,h,j,dk]
// attn is fp32 (converted to bf16 in-register); v stored transposed [b,h,dk,s].
// grid (1, S/64, B*H); K = S = 512.
// ---------------------------------------------------------------------------
__global__ void gemm_av(const float* __restrict__ attn, const bf16* __restrict__ vT,
                        bf16* __restrict__ Ybf) {
    int bh = blockIdx.z;
    int b = bh >> 4, h = bh & 15;
    const float* Aa = attn + (size_t)bh * PS * PS;
    const bf16* Bv = vT + (size_t)bh * PDK * PS;

    int lane = threadIdx.x & 31;
    int wave = threadIdx.x >> 5;
    int wm = blockIdx.y * 64 + (wave >> 1) * 32;
    int wn = (wave & 1) * 32;
    int rA0 = wm + (lane & 15), rA1 = rA0 + 16;
    int cB0 = wn + (lane & 15), cB1 = cB0 + 16;
    v8f c00 = {}, c01 = {}, c10 = {}, c11 = {};
    for (int k0 = 0; k0 < PS; k0 += 32) {
        if (k0 + 64 <= PS) __builtin_prefetch(Aa + (size_t)rA0 * PS + k0 + 32, 0, 0);
        bf16x16 a0 = frag_a_f32(Aa, PS, rA0, k0, lane);
        bf16x16 a1 = frag_a_f32(Aa, PS, rA1, k0, lane);
        bf16x16 b0 = frag_b_bf16(Bv, PS, cB0, k0, lane);
        bf16x16 b1 = frag_b_bf16(Bv, PS, cB1, k0, lane);
        c00 = wmma_bf16(a0, b0, c00);
        c01 = wmma_bf16(a0, b1, c01);
        c10 = wmma_bf16(a1, b0, c10);
        c11 = wmma_bf16(a1, b1, c11);
    }
    int cc = lane & 15;
    int rb = (lane >> 4) * 8;
    v8f accs[4] = {c00, c01, c10, c11};
    int tms[4] = {wm, wm, wm + 16, wm + 16};
    int tns[4] = {wn, wn + 16, wn, wn + 16};
#pragma unroll
    for (int t = 0; t < 4; ++t) {
#pragma unroll
        for (int j = 0; j < 8; ++j) {
            int s = tms[t] + rb + j;
            int dk = tns[t] + cc;
            Ybf[((size_t)b * PS + s) * PD + h * PDK + dk] = (bf16)accs[t][j];
        }
    }
}

// ---------------------------------------------------------------------------
// Host-side launch
// ---------------------------------------------------------------------------
extern "C" void kernel_launch(void* const* d_in, const int* in_sizes, int n_in,
                              void* d_out, int out_size, void* d_ws, size_t ws_size,
                              hipStream_t stream) {
    const float* Q     = (const float*)d_in[0];
    const float* prev  = (const float*)d_in[1];
    const float* Wq    = (const float*)d_in[2];
    const float* bq    = (const float*)d_in[3];
    const float* Wk    = (const float*)d_in[4];
    const float* bk    = (const float*)d_in[5];
    const float* Wv    = (const float*)d_in[6];
    const float* bv    = (const float*)d_in[7];
    const float* Wg    = (const float*)d_in[8];
    const float* bg    = (const float*)d_in[9];
    const float* Wo    = (const float*)d_in[10];
    const float* bo    = (const float*)d_in[11];
    const float* scale = (const float*)d_in[12];

    float* out    = (float*)d_out;                              // [B,S,D]
    float* attn   = out + (size_t)PB * PS * PD;                 // [B,H,S,S]
    float* scores = attn + (size_t)PB * PH * PS * PS;           // [B,H,S,S]

    // Workspace layout (bf16 regions, ~100 MB total)
    uint8_t* w = (uint8_t*)d_ws;
    bf16* Xbf = (bf16*)w;                 w += (size_t)PM * PD * 2;       // 16 MB
    bf16* WqT = (bf16*)w;                 w += (size_t)PD * PD * 2;       //  2 MB
    bf16* WkT = (bf16*)w;                 w += (size_t)PD * PD * 2;
    bf16* WvT = (bf16*)w;                 w += (size_t)PD * PD * 2;
    bf16* WoT = (bf16*)w;                 w += (size_t)PD * PD * 2;
    bf16* WgT = (bf16*)w;                 w += (size_t)PD * PGN * 2;      // 16 MB
    bf16* qbf = (bf16*)w;                 w += (size_t)PB * PH * PS * PDK * 2;
    bf16* kbf = (bf16*)w;                 w += (size_t)PB * PH * PS * PDK * 2;
    bf16* vT  = (bf16*)w;                 w += (size_t)PB * PH * PDK * PS * 2;
    bf16* Ybf = (bf16*)w;                 w += (size_t)PM * PD * 2;

    // 1) fp32 -> bf16 conversions / weight transposes
    {
        int n = PM * PD;
        cvt_f32_bf16<<<dim3((n + 255) / 256), dim3(256), 0, stream>>>(Q, Xbf, n);
        int nw = PD * PD;
        transpose_cvt<<<dim3((nw + 255) / 256), dim3(256), 0, stream>>>(Wq, WqT, PD, PD);
        transpose_cvt<<<dim3((nw + 255) / 256), dim3(256), 0, stream>>>(Wk, WkT, PD, PD);
        transpose_cvt<<<dim3((nw + 255) / 256), dim3(256), 0, stream>>>(Wv, WvT, PD, PD);
        transpose_cvt<<<dim3((nw + 255) / 256), dim3(256), 0, stream>>>(Wo, WoT, PD, PD);
        int ng = PD * PGN;
        transpose_cvt<<<dim3((ng + 255) / 256), dim3(256), 0, stream>>>(Wg, WgT, PD, PGN);
    }

    dim3 blk(256);
    // 2) QKV projections (async-LDS staged bf16 WMMA, f32 accumulate)
    gemm_proj<0><<<dim3(PD / TN, PM / TM), blk, 0, stream>>>(Xbf, WqT, bq, nullptr, qbf, PD);
    gemm_proj<0><<<dim3(PD / TN, PM / TM), blk, 0, stream>>>(Xbf, WkT, bk, nullptr, kbf, PD);
    gemm_proj<1><<<dim3(PD / TN, PM / TM), blk, 0, stream>>>(Xbf, WvT, bv, nullptr, vT, PD);
    // 3) Gating projection -> attn region (holds g until softmax_gate)
    gemm_proj<2><<<dim3(PGN / TN, PM / TM), blk, 0, stream>>>(Xbf, WgT, bg, attn, nullptr, PD);
    // 4) scores = scale * q k^T + prev
    gemm_scores<<<dim3(PS / 64, PS / 64, PB * PH), dim3(128), 0, stream>>>(qbf, kbf, prev, scale, scores);
    // 5) attn = g * softmax(scores)
    softmax_gate<<<dim3(PB * PH * PS), dim3(256), 0, stream>>>(scores, attn);
    // 6) Y = attn @ v  (per head), gathered as [B,S,D] bf16
    gemm_av<<<dim3(1, PS / 64, PB * PH), dim3(128), 0, stream>>>(attn, vT, Ybf);
    // 7) out = Y @ Wo + bo
    gemm_proj<3><<<dim3(PD / TN, PM / TM), blk, 0, stream>>>(Ybf, WoT, bo, out, nullptr, PD);
}
